// PointVoxelXCorr_39127152067009
// MI455X (gfx1250) — matrix operand order repaired and checked by reading, not compiled, not run on hardware
//
#include <hip/hip_runtime.h>
#include <hip/hip_bf16.h>
#include <math.h>

typedef __attribute__((ext_vector_type(16))) __bf16 v16bf;
typedef __attribute__((ext_vector_type(8)))  __bf16 v8bf;
typedef __attribute__((ext_vector_type(8)))  float  v8f;

#define B_    8
#define N1_   2048
#define N2_   512
#define D_    128
#define KNN_  32
#define FC_   192
#define KPAD_ 96   // 81 voxel channels padded to 96 (3 x K=32)

// ---------------- helpers ----------------

__device__ __forceinline__ unsigned short f2bf_bits(float x){
  union { float f; unsigned int u; } c; c.f = x;
  unsigned int r = c.u + 0x7FFFu + ((c.u >> 16) & 1u);
  return (unsigned short)(r >> 16);
}

// A-matrix (MxK, 16x32 bf16) wave32 layout:
//  lanes 0-15 : row M=lane, K=[kb,kb+16)
//  lanes16-31 : row M=lane-16, K=[kb+16,kb+32)
__device__ __forceinline__ v16bf load_a_frag(const unsigned short* row, int kbase, int lane){
  int h = (lane >> 4) & 1;
  v8bf lo = *reinterpret_cast<const v8bf*>(row + kbase + 16*h);
  v8bf hi = *reinterpret_cast<const v8bf*>(row + kbase + 16*h + 8);
  v16bf a;
  #pragma unroll
  for (int i = 0; i < 8; ++i){ a[i] = lo[i]; a[i+8] = hi[i]; }
  return a;
}

// B-matrix (KxN, 32x16 bf16) wave32 layout:
//  lanes 0-15 : col N=lane, K=[kb,kb+16) ; lanes16-31 : col N=lane-16, K=[kb+16,kb+32)
__device__ __forceinline__ v16bf load_b_frag(const unsigned short* row, int kbase, int lane){
  int h = (lane >> 4) & 1;
  v8bf lo = *reinterpret_cast<const v8bf*>(row + kbase + 16*h);
  v8bf hi = *reinterpret_cast<const v8bf*>(row + kbase + 16*h + 8);
  v16bf b;
  #pragma unroll
  for (int i = 0; i < 8; ++i){ b[i] = lo[i]; b[i+8] = hi[i]; }
  return b;
}

// ---------------- pack kernels ----------------

// src [B][dchan][npts] -> dst [B][npts][dchan] bf16 (K-contiguous rows), optional scale
__global__ void k_pack_fmap(const float* __restrict__ src, unsigned short* __restrict__ dst,
                            int dchan, int npts, float scale){
  int i = blockIdx.x * blockDim.x + threadIdx.x;
  int total = B_ * dchan * npts;
  if (i >= total) return;
  int d = i % dchan; int n = (i / dchan) % npts; int b = i / (dchan * npts);
  dst[i] = f2bf_bits(src[((size_t)b * dchan + d) * npts + n] * scale);
}

// W1 [128][81] -> W1p [128][96] bf16, zero padded
__global__ void k_pack_w1(const float* __restrict__ W1, unsigned short* __restrict__ W1p){
  int i = blockIdx.x * blockDim.x + threadIdx.x;
  if (i >= 128 * KPAD_) return;
  int c = i % KPAD_; int o = i / KPAD_;
  W1p[i] = f2bf_bits(c < 81 ? W1[o * 81 + c] : 0.0f);
}

__global__ void k_pack_w(const float* __restrict__ W, unsigned short* __restrict__ Wb, int n){
  int i = blockIdx.x * blockDim.x + threadIdx.x;
  if (i < n) Wb[i] = f2bf_bits(W[i]);
}

__global__ void k_zero(float* __restrict__ p, int n){
  int i = blockIdx.x * blockDim.x + threadIdx.x;
  if (i < n) p[i] = 0.0f;
}

// ---------------- corr = f1^T f2 / sqrt(D) via WMMA ----------------
// one wave -> one 16(n) x 64(m) strip: 4 B-tiles reuse each A fragment.
// scale 1/sqrt(128) pre-folded into f1t.
__global__ void k_corr(const unsigned short* __restrict__ f1t,
                       const unsigned short* __restrict__ f2t,
                       float* __restrict__ corr){
  int gw   = blockIdx.x * (blockDim.x >> 5) + (threadIdx.x >> 5);
  int lane = threadIdx.x & 31;
  int mt4 = gw & 7;            // 8 strips of 64 over N2
  int nt  = (gw >> 3) & 127;   // N1/16
  int b   = gw >> 10;
  int lr  = lane & 15;
  const unsigned short* Arow = f1t + (size_t)(b * N1_ + nt * 16 + lr) * D_;
  const unsigned short* Brow0 = f2t + (size_t)(b * N2_ + mt4 * 64 + lr) * D_;
  v8f acc[4] = {};
  #pragma unroll
  for (int k = 0; k < D_; k += 32){
    v16bf a = load_a_frag(Arow, k, lane);
    #pragma unroll
    for (int t = 0; t < 4; ++t){
      v16bf bb = load_b_frag(Brow0 + (size_t)(t * 16) * D_, k, lane);
      acc[t] = __builtin_amdgcn_wmma_f32_16x16x32_bf16(false, a, false, bb, (short)0, acc[t], false, false);
    }
  }
  int h = lane >> 4;
  int row0 = nt * 16 + 8 * h;
  float* out = corr + (size_t)b * N1_ * N2_;
  #pragma unroll
  for (int t = 0; t < 4; ++t){
    int col = mt4 * 64 + t * 16 + lr;
    #pragma unroll
    for (int r = 0; r < 8; ++r)
      out[(size_t)(row0 + r) * N2_ + col] = acc[t][r];
  }
}

// ---------------- voxel scatter (LDS atomics), write vol as bf16 B-rows ----------------
__global__ void k_voxel(const float* __restrict__ corr,
                        const float* __restrict__ coords,
                        const float* __restrict__ coords2,
                        unsigned short* __restrict__ volb){
  __shared__ float s_add[8][81];
  __shared__ float s_cnt[8][81];
  int wid = threadIdx.x >> 5, lane = threadIdx.x & 31;
  int gp = blockIdx.x * 8 + wid;           // (b, n) point id
  int n = gp & (N1_ - 1), b = gp / N1_;
  for (int j = lane; j < 81; j += 32){ s_add[wid][j] = 0.0f; s_cnt[wid][j] = 0.0f; }
  __syncthreads();
  float cx = coords[(size_t)gp * 3 + 0];
  float cy = coords[(size_t)gp * 3 + 1];
  float cz = coords[(size_t)gp * 3 + 2];
  const float* crow = corr + (size_t)gp * N2_;
  __builtin_prefetch(crow + lane, 0, 0);
  for (int m = lane; m < N2_; m += 32){
    if (m + 32 < N2_) __builtin_prefetch(crow + m + 32, 0, 0);
    float dx = coords2[((size_t)b * N2_ + m) * 3 + 0] - cx;
    float dy = coords2[((size_t)b * N2_ + m) * 3 + 1] - cy;
    float dz = coords2[((size_t)b * N2_ + m) * 3 + 2] - cz;
    float cv = crow[m];
    float r = 0.25f;
    #pragma unroll
    for (int lvl = 0; lvl < 3; ++lvl){
      float ix = rintf(dx / r), iy = rintf(dy / r), iz = rintf(dz / r);
      if (fabsf(ix) <= 1.0f && fabsf(iy) <= 1.0f && fabsf(iz) <= 1.0f){
        int idx = lvl * 27 + ((int)ix + 1) * 9 + ((int)iy + 1) * 3 + ((int)iz + 1);
        atomicAdd(&s_add[wid][idx], cv);
        atomicAdd(&s_cnt[wid][idx], 1.0f);
      }
      r *= 2.0f;
    }
  }
  __syncthreads();
  unsigned short* vrow = volb + (size_t)gp * KPAD_;
  for (int j = lane; j < KPAD_; j += 32){
    float v = 0.0f;
    if (j < 81) v = s_add[wid][j] / fmaxf(s_cnt[wid][j], 1.0f);
    vrow[j] = f2bf_bits(v);
  }
}

// ---------------- h1 = W1p x vol + b1 via WMMA (K=96), 2 N-tiles/wave ----------------
__global__ void k_h1(const unsigned short* __restrict__ W1p,
                     const unsigned short* __restrict__ volb,
                     const float* __restrict__ b1,
                     float* __restrict__ h1){
  int gw   = blockIdx.x * (blockDim.x >> 5) + (threadIdx.x >> 5);
  int lane = threadIdx.x & 31;
  int nt2 = gw & 63; int ot = (gw >> 6) & 7; int b = gw >> 9;
  int lr = lane & 15;
  const unsigned short* Arow = W1p + (size_t)(ot * 16 + lr) * KPAD_;
  const unsigned short* Brow0 = volb + (size_t)(b * N1_ + nt2 * 32 + lr) * KPAD_;
  v8f acc[2] = {};
  #pragma unroll
  for (int k = 0; k < KPAD_; k += 32){
    v16bf a = load_a_frag(Arow, k, lane);
    #pragma unroll
    for (int t = 0; t < 2; ++t){
      v16bf bb = load_b_frag(Brow0 + (size_t)(t * 16) * KPAD_, k, lane);
      acc[t] = __builtin_amdgcn_wmma_f32_16x16x32_bf16(false, a, false, bb, (short)0, acc[t], false, false);
    }
  }
  int h = lane >> 4; int o0 = ot * 16 + 8 * h;
  #pragma unroll
  for (int t = 0; t < 2; ++t){
    int n = nt2 * 32 + t * 16 + lr;
    #pragma unroll
    for (int r = 0; r < 8; ++r){
      int o = o0 + r;
      h1[(size_t)(b * 128 + o) * N1_ + n] = acc[t][r] + b1[o];
    }
  }
}

// ---------------- GN stats for h1: one block per (b,group) ----------------
__global__ void k_stats1(const float* __restrict__ h1, float* __restrict__ st1){
  __shared__ float rs[256], rs2[256];
  int b = blockIdx.x >> 3, g = blockIdx.x & 7;
  const float* base = h1 + (size_t)(b * 128 + g * 16) * N1_;
  float s = 0.0f, s2 = 0.0f;
  for (int i = threadIdx.x; i < 16 * N1_; i += 256){ float x = base[i]; s += x; s2 += x * x; }
  rs[threadIdx.x] = s; rs2[threadIdx.x] = s2; __syncthreads();
  for (int st = 128; st > 0; st >>= 1){
    if (threadIdx.x < st){ rs[threadIdx.x] += rs[threadIdx.x + st]; rs2[threadIdx.x] += rs2[threadIdx.x + st]; }
    __syncthreads();
  }
  if (threadIdx.x == 0){
    float inv = 1.0f / (16.0f * N1_);
    float mean = rs[0] * inv;
    float var = rs2[0] * inv - mean * mean;
    st1[blockIdx.x * 2 + 0] = mean;
    st1[blockIdx.x * 2 + 1] = rsqrtf(var + 1e-5f);
  }
}

// ---------------- normalize+PReLU h1 -> bf16 B-rows [b][n][128] ----------------
__global__ void k_norm1(const float* __restrict__ h1, const float* __restrict__ st1,
                        const float* __restrict__ g1, const float* __restrict__ be1,
                        const float* __restrict__ a1p, unsigned short* __restrict__ h1n){
  int i = blockIdx.x * blockDim.x + threadIdx.x;
  if (i >= B_ * 128 * N1_) return;
  int n = i & (N1_ - 1); int c = (i / N1_) & 127; int b = i / (N1_ * 128);
  int g = c >> 4;
  float mean = st1[(b * 8 + g) * 2 + 0];
  float rstd = st1[(b * 8 + g) * 2 + 1];
  float y = (h1[i] - mean) * rstd * g1[c] + be1[c];
  float a1 = a1p[0];
  y = (y >= 0.0f) ? y : a1 * y;
  h1n[((size_t)b * N1_ + n) * 128 + c] = f2bf_bits(y);
}

// ---------------- top-32 nearest neighbors per point ----------------
__global__ void k_knn(const float* __restrict__ coords, const float* __restrict__ coords2,
                      int* __restrict__ nbr){
  __shared__ float s_c2[N2_ * 3];
  int b = blockIdx.x >> 3;
  int n = ((blockIdx.x & 7) << 8) + threadIdx.x;
  for (int j = threadIdx.x; j < N2_ * 3; j += 256) s_c2[j] = coords2[(size_t)b * N2_ * 3 + j];
  __syncthreads();
  int gp = b * N1_ + n;
  float cx = coords[(size_t)gp * 3 + 0];
  float cy = coords[(size_t)gp * 3 + 1];
  float cz = coords[(size_t)gp * 3 + 2];
  float bd[KNN_]; int bi[KNN_];
  #pragma unroll
  for (int m = 0; m < KNN_; ++m){
    float dx = s_c2[m * 3 + 0] - cx, dy = s_c2[m * 3 + 1] - cy, dz = s_c2[m * 3 + 2] - cz;
    bd[m] = dx * dx + dy * dy + dz * dz; bi[m] = m;
  }
  int wi = 0; float wv = bd[0];
  #pragma unroll
  for (int m = 1; m < KNN_; ++m) if (bd[m] > wv){ wv = bd[m]; wi = m; }
  for (int m = KNN_; m < N2_; ++m){
    float dx = s_c2[m * 3 + 0] - cx, dy = s_c2[m * 3 + 1] - cy, dz = s_c2[m * 3 + 2] - cz;
    float d = dx * dx + dy * dy + dz * dz;
    if (d < wv){
      bd[wi] = d; bi[wi] = m;
      wv = bd[0]; wi = 0;
      #pragma unroll
      for (int j = 1; j < KNN_; ++j) if (bd[j] > wv){ wv = bd[j]; wi = j; }
    }
  }
  #pragma unroll
  for (int k = 0; k < KNN_; ++k) nbr[(size_t)gp * KNN_ + k] = bi[k];
}

// ---------------- knn conv group stats (pass 1, atomics) ----------------
__global__ void k_knn_stats(const float* __restrict__ corr, const float* __restrict__ coords,
                            const float* __restrict__ coords2, const int* __restrict__ nbr,
                            const float* __restrict__ Wk, const float* __restrict__ bk,
                            float* __restrict__ stk){
  __shared__ float s_wk[256], s_bk[64], s_g[8], s_g2[8];
  int tid = threadIdx.x;
  if (tid < 256) s_wk[tid] = Wk[tid];
  if (tid < 64)  s_bk[tid] = bk[tid];
  if (tid < 8){ s_g[tid] = 0.0f; s_g2[tid] = 0.0f; }
  __syncthreads();
  int i = blockIdx.x * 256 + tid;                 // over B*N1*K
  int n = (i / KNN_) & (N1_ - 1);
  int b = i / (KNN_ * N1_);
  int gp = b * N1_ + n;
  int m = nbr[i];
  float e0 = corr[(size_t)gp * N2_ + m];
  float e1 = coords2[((size_t)b * N2_ + m) * 3 + 0] - coords[(size_t)gp * 3 + 0];
  float e2 = coords2[((size_t)b * N2_ + m) * 3 + 1] - coords[(size_t)gp * 3 + 1];
  float e3 = coords2[((size_t)b * N2_ + m) * 3 + 2] - coords[(size_t)gp * 3 + 2];
  float gs[8], gs2[8];
  #pragma unroll
  for (int g = 0; g < 8; ++g){ gs[g] = 0.0f; gs2[g] = 0.0f; }
  #pragma unroll
  for (int c = 0; c < 64; ++c){
    float h = s_wk[c*4+0]*e0 + s_wk[c*4+1]*e1 + s_wk[c*4+2]*e2 + s_wk[c*4+3]*e3 + s_bk[c];
    gs[c >> 3] += h; gs2[c >> 3] += h * h;
  }
  #pragma unroll
  for (int g = 0; g < 8; ++g){ atomicAdd(&s_g[g], gs[g]); atomicAdd(&s_g2[g], gs2[g]); }
  __syncthreads();
  if (tid < 8){
    atomicAdd(&stk[(b * 8 + tid) * 2 + 0], s_g[tid]);
    atomicAdd(&stk[(b * 8 + tid) * 2 + 1], s_g2[tid]);
  }
}

__global__ void k_statsk_fin(const float* __restrict__ stk, float* __restrict__ stkf){
  int i = threadIdx.x;  // 64 = B*8
  if (i >= 64) return;
  float inv = 1.0f / (8.0f * N1_ * KNN_);
  float mean = stk[i * 2 + 0] * inv;
  float var  = stk[i * 2 + 1] * inv - mean * mean;
  stkf[i * 2 + 0] = mean;
  stkf[i * 2 + 1] = rsqrtf(var + 1e-5f);
}

// ---------------- knn conv pass 2: norm + PReLU + max over k -> bf16 rows ----------------
__global__ void k_knn_max(const float* __restrict__ corr, const float* __restrict__ coords,
                          const float* __restrict__ coords2, const int* __restrict__ nbr,
                          const float* __restrict__ Wk, const float* __restrict__ bk,
                          const float* __restrict__ gk, const float* __restrict__ bek,
                          const float* __restrict__ akp, const float* __restrict__ stkf,
                          unsigned short* __restrict__ hkb){
  __shared__ float s_e[4][KNN_][4];
  int tid = threadIdx.x;
  int pl = tid >> 6;          // 4 points per block
  int c  = tid & 63;
  int gp = blockIdx.x * 4 + pl;
  int n = gp & (N1_ - 1), b = gp / N1_;
  if (c < KNN_){
    int m = nbr[(size_t)gp * KNN_ + c];
    s_e[pl][c][0] = corr[(size_t)gp * N2_ + m];
    s_e[pl][c][1] = coords2[((size_t)b * N2_ + m) * 3 + 0] - coords[(size_t)gp * 3 + 0];
    s_e[pl][c][2] = coords2[((size_t)b * N2_ + m) * 3 + 1] - coords[(size_t)gp * 3 + 1];
    s_e[pl][c][3] = coords2[((size_t)b * N2_ + m) * 3 + 2] - coords[(size_t)gp * 3 + 2];
  }
  __syncthreads();
  float w0 = Wk[c*4+0], w1 = Wk[c*4+1], w2 = Wk[c*4+2], w3 = Wk[c*4+3];
  float bc = bk[c], gc = gk[c], bec = bek[c], ak = akp[0];
  int g = c >> 3;
  float mean = stkf[(b * 8 + g) * 2 + 0];
  float rstd = stkf[(b * 8 + g) * 2 + 1];
  float acc = -3.4e38f;
  #pragma unroll
  for (int k = 0; k < KNN_; ++k){
    float h = w0*s_e[pl][k][0] + w1*s_e[pl][k][1] + w2*s_e[pl][k][2] + w3*s_e[pl][k][3] + bc;
    h = (h - mean) * rstd * gc + bec;
    h = (h >= 0.0f) ? h : ak * h;
    acc = fmaxf(acc, h);
  }
  hkb[((size_t)b * N1_ + n) * 64 + c] = f2bf_bits(acc);
}

// ---------------- final: out = W2 x h1n + Wo x hk + (b2+bo) via WMMA, 2 N-tiles/wave ----------------
__global__ void k_final(const unsigned short* __restrict__ W2b,
                        const unsigned short* __restrict__ h1n,
                        const unsigned short* __restrict__ Wob,
                        const unsigned short* __restrict__ hkb,
                        const float* __restrict__ b2, const float* __restrict__ bo,
                        float* __restrict__ out){
  int gw   = blockIdx.x * (blockDim.x >> 5) + (threadIdx.x >> 5);
  int lane = threadIdx.x & 31;
  int nt2 = gw & 63; int ot = (gw >> 6) % 12; int b = gw / (64 * 12);
  int lr = lane & 15;
  const unsigned short* A1 = W2b + (size_t)(ot * 16 + lr) * 128;
  const unsigned short* B1 = h1n + (size_t)(b * N1_ + nt2 * 32 + lr) * 128;
  const unsigned short* A2 = Wob + (size_t)(ot * 16 + lr) * 64;
  const unsigned short* B2 = hkb + (size_t)(b * N1_ + nt2 * 32 + lr) * 64;
  v8f acc[2] = {};
  #pragma unroll
  for (int k = 0; k < 128; k += 32){
    v16bf a = load_a_frag(A1, k, lane);
    #pragma unroll
    for (int t = 0; t < 2; ++t){
      v16bf bb = load_b_frag(B1 + (size_t)(t * 16) * 128, k, lane);
      acc[t] = __builtin_amdgcn_wmma_f32_16x16x32_bf16(false, a, false, bb, (short)0, acc[t], false, false);
    }
  }
  #pragma unroll
  for (int k = 0; k < 64; k += 32){
    v16bf a = load_a_frag(A2, k, lane);
    #pragma unroll
    for (int t = 0; t < 2; ++t){
      v16bf bb = load_b_frag(B2 + (size_t)(t * 16) * 64, k, lane);
      acc[t] = __builtin_amdgcn_wmma_f32_16x16x32_bf16(false, a, false, bb, (short)0, acc[t], false, false);
    }
  }
  int h = lane >> 4; int o0 = ot * 16 + 8 * h;
  #pragma unroll
  for (int t = 0; t < 2; ++t){
    int n = nt2 * 32 + t * 16 + lr;
    #pragma unroll
    for (int r = 0; r < 8; ++r){
      int o = o0 + r;
      out[(size_t)(b * FC_ + o) * N1_ + n] = acc[t][r] + b2[o] + bo[o];
    }
  }
}

// ---------------- workspace layout (bytes, 256-aligned) ----------------
static constexpr size_t OFF_F1T  = 0;                        // 4,194,304  bf16 [B][N1][D] (pre-scaled)
static constexpr size_t OFF_F2T  = OFF_F1T  + 4194304;       // 1,048,576  bf16 [B][N2][D]
static constexpr size_t OFF_W1P  = OFF_F2T  + 1048576;       // 24,576     bf16 [128][96]
static constexpr size_t OFF_W2B  = OFF_W1P  + 24576;         // 49,152     bf16 [192][128]
static constexpr size_t OFF_WOB  = OFF_W2B  + 49152;         // 24,576     bf16 [192][64]
static constexpr size_t OFF_CORR = OFF_WOB  + 24576;         // 33,554,432 f32  [B][N1][N2]
static constexpr size_t OFF_VOL  = OFF_CORR + 33554432;      // 3,145,728  bf16 [B][N1][96]
static constexpr size_t OFF_H1   = OFF_VOL  + 3145728;       // 8,388,608  f32  [B][128][N1]
static constexpr size_t OFF_ST1  = OFF_H1   + 8388608;       // 512
static constexpr size_t OFF_H1N  = OFF_ST1  + 512;           // 4,194,304  bf16 [B][N1][128]
static constexpr size_t OFF_NBR  = OFF_H1N  + 4194304;       // 2,097,152  i32  [B][N1][K]
static constexpr size_t OFF_STK  = OFF_NBR  + 2097152;       // 512 raw sums
static constexpr size_t OFF_STKF = OFF_STK  + 512;           // 512 finalized
static constexpr size_t OFF_HKB  = OFF_STKF + 512;           // 2,097,152  bf16 [B][N1][64]

extern "C" void kernel_launch(void* const* d_in, const int* in_sizes, int n_in,
                              void* d_out, int out_size, void* d_ws, size_t ws_size,
                              hipStream_t stream) {
  (void)in_sizes; (void)n_in; (void)out_size; (void)ws_size;
  const float* coords  = (const float*)d_in[0];
  const float* coords2 = (const float*)d_in[1];
  const float* fmap1   = (const float*)d_in[2];
  const float* fmap2   = (const float*)d_in[3];
  const float* W1  = (const float*)d_in[4];
  const float* b1  = (const float*)d_in[5];
  const float* g1  = (const float*)d_in[6];
  const float* be1 = (const float*)d_in[7];
  const float* a1  = (const float*)d_in[8];
  const float* W2  = (const float*)d_in[9];
  const float* b2  = (const float*)d_in[10];
  const float* Wk  = (const float*)d_in[11];
  const float* bk  = (const float*)d_in[12];
  const float* gk  = (const float*)d_in[13];
  const float* bek = (const float*)d_in[14];
  const float* ak  = (const float*)d_in[15];
  const float* Wo  = (const float*)d_in[16];
  const float* bo  = (const float*)d_in[17];
  float* out = (float*)d_out;

  char* ws = (char*)d_ws;
  unsigned short* f1t  = (unsigned short*)(ws + OFF_F1T);
  unsigned short* f2t  = (unsigned short*)(ws + OFF_F2T);
  unsigned short* W1p  = (unsigned short*)(ws + OFF_W1P);
  unsigned short* W2b  = (unsigned short*)(ws + OFF_W2B);
  unsigned short* Wob  = (unsigned short*)(ws + OFF_WOB);
  float*          corr = (float*)(ws + OFF_CORR);
  unsigned short* volb = (unsigned short*)(ws + OFF_VOL);
  float*          h1   = (float*)(ws + OFF_H1);
  float*          st1  = (float*)(ws + OFF_ST1);
  unsigned short* h1n  = (unsigned short*)(ws + OFF_H1N);
  int*            nbr  = (int*)(ws + OFF_NBR);
  float*          stk  = (float*)(ws + OFF_STK);
  float*          stkf = (float*)(ws + OFF_STKF);
  unsigned short* hkb  = (unsigned short*)(ws + OFF_HKB);

  const float inv_sqrt_d = 0.08838834764831845f; // 1/sqrt(128), folded into f1t

  // pack
  k_pack_fmap<<<(B_*D_*N1_ + 255)/256, 256, 0, stream>>>(fmap1, f1t, D_, N1_, inv_sqrt_d);
  k_pack_fmap<<<(B_*D_*N2_ + 255)/256, 256, 0, stream>>>(fmap2, f2t, D_, N2_, 1.0f);
  k_pack_w1  <<<(128*KPAD_ + 255)/256, 256, 0, stream>>>(W1, W1p);
  k_pack_w   <<<(FC_*128 + 255)/256,   256, 0, stream>>>(W2, W2b, FC_*128);
  k_pack_w   <<<(FC_*64  + 255)/256,   256, 0, stream>>>(Wo, Wob, FC_*64);
  k_zero     <<<1, 128, 0, stream>>>(stk, 128);

  // corr GEMM: B * 128 (n-tiles) * 8 (64-wide m strips) waves, 8 waves/block
  k_corr<<<B_*128*8/8, 256, 0, stream>>>(f1t, f2t, corr);

  // voxel scatter + vol pack
  k_voxel<<<B_*N1_/8, 256, 0, stream>>>(corr, coords, coords2, volb);

  // vol_conv stage 1 + GN + PReLU
  k_h1    <<<B_*8*64/8, 256, 0, stream>>>(W1p, volb, b1, h1);
  k_stats1<<<B_*8, 256, 0, stream>>>(h1, st1);
  k_norm1 <<<(B_*128*N1_ + 255)/256, 256, 0, stream>>>(h1, st1, g1, be1, a1, h1n);

  // knn branch
  k_knn       <<<B_*N1_/256, 256, 0, stream>>>(coords, coords2, nbr);
  k_knn_stats <<<B_*N1_*KNN_/256, 256, 0, stream>>>(corr, coords, coords2, nbr, Wk, bk, stk);
  k_statsk_fin<<<1, 64, 0, stream>>>(stk, stkf);
  k_knn_max   <<<B_*N1_/4, 256, 0, stream>>>(corr, coords, coords2, nbr, Wk, bk, gk, bek, ak, stkf, hkb);

  // final fused GEMM: out = W2*h1n + Wo*hk + bias
  k_final<<<B_*12*64/8, 256, 0, stream>>>(W2b, h1n, Wob, hkb, b2, bo, out);
}
